// MSAAttention_9706626089896
// MI455X (gfx1250) — matrix operand order repaired and checked by reading, not compiled, not run on hardware
//
#include <hip/hip_runtime.h>
#include <hip/hip_bf16.h>

// ---------- types ----------
typedef float v8f  __attribute__((ext_vector_type(8)));
typedef __bf16 v16bf __attribute__((ext_vector_type(16)));
typedef unsigned v4u __attribute__((ext_vector_type(4)));
typedef int v4i __attribute__((ext_vector_type(4)));
typedef int v8i __attribute__((ext_vector_type(8)));

union ABFrag { v16bf v; unsigned u[8]; };

__device__ __forceinline__ unsigned short bf16raw(float f) {
    union { float f; unsigned u; } x; x.f = f;
    unsigned r = x.u + 0x7FFFu + ((x.u >> 16) & 1u);   // round-to-nearest-even
    return (unsigned short)(r >> 16);
}
__device__ __forceinline__ float bf16tof(unsigned short h) {
    union { unsigned u; float f; } x; x.u = ((unsigned)h) << 16; return x.f;
}

#define S_   128
#define R_   384
#define DIN  256
#define DPAIR 128
#define H_   8
#define HD   32
#define TOT  256
#define SR   (S_ * R_)        // 49152
#define QSCALE 0.17677669529663687f  // 32^-0.5
#define EPS  1e-5f

#if defined(__HIP_DEVICE_COMPILE__) && __has_builtin(__builtin_amdgcn_tensor_load_to_lds) && __has_builtin(__builtin_amdgcn_s_wait_tensorcnt)
#define USE_TDM 1
#else
#define USE_TDM 0
#endif

#define LAS 20   // padded LDS row stride in dwords (32 bf16 = 16 dwords + 4 pad)

#if USE_TDM
// Issue one TDM load: 64 rows x 32 bf16 tile out of a [rows x 256] bf16
// row-major matrix, into LDS at ldsOff with 4-dword padding after each
// 16-dword row (=> LAS stride). 2-D tile: groups 2/3 zero.
__device__ __forceinline__ void tdm_load_tile(unsigned ldsOff, unsigned long long ga,
                                              unsigned rows) {
    v4u g0; v8i g1; v4i gz; gz[0] = 0; gz[1] = 0; gz[2] = 0; gz[3] = 0;
    g0[0] = 1u;                                             // count=1, user desc
    g0[1] = ldsOff;                                         // lds_addr (bytes)
    g0[2] = (unsigned)ga;                                   // global_addr[31:0]
    g0[3] = (unsigned)((ga >> 32) & 0x01FFFFFFu) | 0x80000000u; // addr[56:32] | type=2
    // data_size=1(2B), pad_enable, pad_interval=3(16 dw), pad_amount=3(4 dw)
    g1[0] = (int)((1u << 16) | (1u << 20) | (3u << 22) | (3u << 25));
    g1[1] = (int)((256u & 0xFFFFu) << 16);                  // tensor_dim0=256 (lo16)
    g1[2] = (int)((256u >> 16) | ((rows & 0xFFFFu) << 16)); // dim0 hi | tensor_dim1 lo
    g1[3] = (int)((rows >> 16) | (32u << 16));              // dim1 hi | tile_dim0=32
    g1[4] = 64;                                             // tile_dim1=64, tile_dim2=0
    g1[5] = 256;                                            // tensor_dim0_stride lo32
    g1[6] = 0;                                              // stride hi | dim1_stride lo
    g1[7] = 0;
#if defined(__clang_major__) && __clang_major__ >= 23
    v8i gz8; for (int i = 0; i < 8; ++i) gz8[i] = 0;
    __builtin_amdgcn_tensor_load_to_lds(g0, g1, gz, gz, gz8, 0);
#else
    __builtin_amdgcn_tensor_load_to_lds(g0, g1, gz, gz, 0);
#endif
}
#endif

// ---------------------------------------------------------------------------
// Kernel 1: convert 5 weight matrices (Wq,Wk,Wv,Wg,Wo) fp32 -> bf16
// ---------------------------------------------------------------------------
__global__ __launch_bounds__(256)
void convw_kernel(const float* __restrict__ wq, const float* __restrict__ wk,
                  const float* __restrict__ wv, const float* __restrict__ wg,
                  const float* __restrict__ wo, unsigned short* __restrict__ wbf) {
    int wid = blockIdx.y;
    const float* src = (wid == 0) ? wq : (wid == 1) ? wk : (wid == 2) ? wv
                     : (wid == 3) ? wg : wo;
    int i = blockIdx.x * 256 + threadIdx.x;       // < 65536
    wbf[(long)wid * 65536 + i] = bf16raw(src[i]);
}

// ---------------------------------------------------------------------------
// Kernel 2: layernorm(m) -> bf16 m_n.  grid 49152 x 256
// ---------------------------------------------------------------------------
__global__ __launch_bounds__(256)
void ln_m_kernel(const float* __restrict__ m, const float* __restrict__ w,
                 const float* __restrict__ b, unsigned short* __restrict__ mn) {
    __shared__ float red[256];
    const int t = threadIdx.x;
    const long row = blockIdx.x;
    float x = m[row * DIN + t];
    red[t] = x; __syncthreads();
    for (int s = 128; s > 0; s >>= 1) { if (t < s) red[t] += red[t + s]; __syncthreads(); }
    float mu = red[0] * (1.0f / DIN); __syncthreads();
    float d = x - mu;
    red[t] = d * d; __syncthreads();
    for (int s = 128; s > 0; s >>= 1) { if (t < s) red[t] += red[t + s]; __syncthreads(); }
    float var = red[0] * (1.0f / DIN);
    mn[row * DIN + t] = bf16raw(d * rsqrtf(var + EPS) * w[t] + b[t]);
}

// ---------------------------------------------------------------------------
// Kernel 3: pair bias = einsum(LN(z), Wz) -> f32 bias[h][q][k]
// ---------------------------------------------------------------------------
__global__ __launch_bounds__(128)
void pair_bias_kernel(const float* __restrict__ z, const float* __restrict__ lnw,
                      const float* __restrict__ lnb, const float* __restrict__ Wz,
                      float* __restrict__ biasb) {
    __shared__ float red[128];
    const int t = threadIdx.x;
    const long rowqk = blockIdx.x;               // q*384 + k
    float x = z[rowqk * DPAIR + t];
    red[t] = x; __syncthreads();
    for (int s = 64; s > 0; s >>= 1) { if (t < s) red[t] += red[t + s]; __syncthreads(); }
    float mu = red[0] * (1.0f / DPAIR); __syncthreads();
    float d = x - mu;
    red[t] = d * d; __syncthreads();
    for (int s = 64; s > 0; s >>= 1) { if (t < s) red[t] += red[t + s]; __syncthreads(); }
    float var = red[0] * (1.0f / DPAIR); __syncthreads();
    float zn = d * rsqrtf(var + EPS) * lnw[t] + lnb[t];
    for (int h = 0; h < H_; ++h) {
        red[t] = zn * Wz[h * DPAIR + t]; __syncthreads();
        for (int s = 64; s > 0; s >>= 1) { if (t < s) red[t] += red[t + s]; __syncthreads(); }
        if (t == 0) biasb[(long)h * (R_ * R_) + rowqk] = red[0];
        __syncthreads();
    }
}

// ---------------------------------------------------------------------------
// Shared GEMM body: 64x64 C-tile, K=256, bf16 WMMA, double-buffered LDS
// tiles fed by the Tensor Data Mover (fallback: synchronous staging).
// Both A [Mrows x 256] and B [Nrows x 256] are row-major bf16, K contiguous.
// ---------------------------------------------------------------------------
#define GEMM_PROLOG() \
    const int tid = threadIdx.x, lane = tid & 31, widx = tid >> 5;             \
    const int half = lane >> 4, mr = lane & 15;                                \
    const int mt0 = (widx * 2) >> 2, nt0 = (widx * 2) & 3;                     \
    const int mt1 = (widx * 2 + 1) >> 2, nt1 = (widx * 2 + 1) & 3;             \
    v8f acc0 = {}, acc1 = {};

#if USE_TDM
#define GEMM_CORE(Aptr, Arows, Bptr, Brows, rbase, nbase)                      \
    {                                                                          \
        const unsigned long long gaA0 =                                        \
            (unsigned long long)(uintptr_t)(Aptr) + (unsigned long long)(rbase) * 512ull; \
        const unsigned long long gaB0 =                                        \
            (unsigned long long)(uintptr_t)(Bptr) + (unsigned long long)(nbase) * 512ull; \
        if (widx == 0) {                                                       \
            tdm_load_tile((unsigned)(uintptr_t)&lA[0][0], gaA0, (Arows));      \
            tdm_load_tile((unsigned)(uintptr_t)&lB[0][0], gaB0, (Brows));      \
        }                                                                      \
        for (int ks = 0; ks < 8; ++ks) {                                       \
            if (widx == 0) {                                                   \
                if (ks + 1 < 8) {                                              \
                    int nb = (ks + 1) & 1;                                     \
                    tdm_load_tile((unsigned)(uintptr_t)&lA[nb][0],             \
                                  gaA0 + (unsigned)(ks + 1) * 64u, (Arows));   \
                    tdm_load_tile((unsigned)(uintptr_t)&lB[nb][0],             \
                                  gaB0 + (unsigned)(ks + 1) * 64u, (Brows));   \
                    __builtin_amdgcn_s_wait_tensorcnt(2);                      \
                } else {                                                       \
                    __builtin_amdgcn_s_wait_tensorcnt(0);                      \
                }                                                              \
            }                                                                  \
            __syncthreads();                                                   \
            const unsigned* cA = &lA[ks & 1][0];                               \
            const unsigned* cB = &lB[ks & 1][0];                               \
            ABFrag a0, a1, b0, b1;                                             \
            _Pragma("unroll")                                                  \
            for (int j = 0; j < 8; ++j) {                                      \
                int ka = (j & 3) + 8 * (j >> 2) + 4 * half;                    \
                int kbw = (j & 3) + 8 * (j >> 2) + 4 * half;                   \
                a0.u[j] = cA[(mt0 * 16 + mr) * LAS + ka];                      \
                b0.u[j] = cB[(nt0 * 16 + mr) * LAS + kbw];                     \
                a1.u[j] = cA[(mt1 * 16 + mr) * LAS + ka];                      \
                b1.u[j] = cB[(nt1 * 16 + mr) * LAS + kbw];                     \
            }                                                                  \
            acc0 = __builtin_amdgcn_wmma_f32_16x16x32_bf16(false, a0.v, false, b0.v, (short)0, acc0, false, false); \
            acc1 = __builtin_amdgcn_wmma_f32_16x16x32_bf16(false, a1.v, false, b1.v, (short)0, acc1, false, false); \
            __syncthreads();                                                   \
        }                                                                      \
    }
#else
#define GEMM_CORE(Aptr, Arows, Bptr, Brows, rbase, nbase)                      \
    {                                                                          \
        const int rowL = tid >> 2, colL = (tid & 3) * 4;                       \
        for (int ks = 0; ks < 8; ++ks) {                                       \
            uint4 va = *(const uint4*)((Aptr) + (long)((rbase) + rowL) * 128 + ks * 16 + colL); \
            uint4 vb = *(const uint4*)((Bptr) + (long)((nbase) + rowL) * 128 + ks * 16 + colL); \
            *(uint4*)&lA[0][rowL * LAS + colL] = va;                           \
            *(uint4*)&lB[0][rowL * LAS + colL] = vb;                           \
            __syncthreads();                                                   \
            const unsigned* cA = &lA[0][0];                                    \
            const unsigned* cB = &lB[0][0];                                    \
            ABFrag a0, a1, b0, b1;                                             \
            _Pragma("unroll")                                                  \
            for (int j = 0; j < 8; ++j) {                                      \
                int ka = (j & 3) + 8 * (j >> 2) + 4 * half;                    \
                int kbw = (j & 3) + 8 * (j >> 2) + 4 * half;                   \
                a0.u[j] = cA[(mt0 * 16 + mr) * LAS + ka];                      \
                b0.u[j] = cB[(nt0 * 16 + mr) * LAS + kbw];                     \
                a1.u[j] = cA[(mt1 * 16 + mr) * LAS + ka];                      \
                b1.u[j] = cB[(nt1 * 16 + mr) * LAS + kbw];                     \
            }                                                                  \
            acc0 = __builtin_amdgcn_wmma_f32_16x16x32_bf16(false, a0.v, false, b0.v, (short)0, acc0, false, false); \
            acc1 = __builtin_amdgcn_wmma_f32_16x16x32_bf16(false, a1.v, false, b1.v, (short)0, acc1, false, false); \
            __syncthreads();                                                   \
        }                                                                      \
    }
#endif

// ---------------------------------------------------------------------------
// Kernel 4: QKVG projection. C[49152, 256] = m_n * W^T. grid (768, 4, 4)
// ---------------------------------------------------------------------------
__global__ __launch_bounds__(256)
void proj_kernel(const unsigned* __restrict__ mn_u, const unsigned* __restrict__ wbf_u,
                 const float* __restrict__ bg,
                 unsigned short* __restrict__ qb, unsigned short* __restrict__ kb,
                 unsigned short* __restrict__ vt, unsigned short* __restrict__ gb) {
    __shared__ unsigned lA[2][64 * LAS];
    __shared__ unsigned lB[2][64 * LAS];
    GEMM_PROLOG();
    const int rbase = blockIdx.x * 64, nbase = blockIdx.y * 64, wid = blockIdx.z;
    const unsigned* Bw = wbf_u + (long)wid * 32768;          // 256*256/2 dwords
    GEMM_CORE(mn_u, (unsigned)SR, Bw, 256u, rbase, nbase);

    auto store_tile = [&](v8f c, int mt, int nt) {
        #pragma unroll
        for (int i = 0; i < 8; ++i) {
            int gm = rbase + mt * 16 + i + 8 * half;         // row sr
            int gn = nbase + nt * 16 + mr;                   // col e
            float v = c[i];
            long o = (long)gm * TOT + gn;
            if (wid == 0)      qb[o] = bf16raw(v * QSCALE);
            else if (wid == 1) kb[o] = bf16raw(v);
            else if (wid == 2) {
                int s = gm / R_, r = gm % R_;                // transpose: vt[s*256+e][r]
                vt[((long)s * TOT + gn) * R_ + r] = bf16raw(v);
            } else {
                float g = 1.0f / (1.0f + __expf(-(v + bg[gn])));
                gb[o] = bf16raw(g);
            }
        }
    };
    store_tile(acc0, mt0, nt0);
    store_tile(acc1, mt1, nt1);
}

// ---------------------------------------------------------------------------
// Kernel 5: attention core. One block per (qtile=64, h, s). grid (6,8,128)
// ---------------------------------------------------------------------------
#define SCS 388    // padded score row stride (f32)
#define PST 400    // padded P row stride (bf16)

__global__ __launch_bounds__(256)
void attn_kernel(const unsigned* __restrict__ qbu, const unsigned* __restrict__ kbu,
                 const unsigned* __restrict__ vtu, const unsigned short* __restrict__ gb,
                 const float* __restrict__ biasb, const float* __restrict__ mask,
                 unsigned short* __restrict__ og) {
    extern __shared__ char smem[];
    float*          sc  = (float*)smem;                                  // 64*SCS f32
    unsigned short* lP  = (unsigned short*)(smem + 64 * SCS * 4);        // 64*PST bf16
    float*          red = (float*)(smem + 64 * SCS * 4 + 64 * PST * 2);  // 256 f32

    const int tid = threadIdx.x, lane = tid & 31, widx = tid >> 5;
    const int half = lane >> 4, mr = lane & 15;
    const int qt = blockIdx.x, h = blockIdx.y, s = blockIdx.z;
    const int q0 = qt * 64;
    const long rowQ0 = (long)s * R_ + q0;

    // ---- Phase 1: 4x24 = 96 WMMA tiles of Q*K^T (K-dim = hd = 32, 1 step)
    for (int t = widx; t < 96; t += 8) {
        int mt = t / 24, nt = t % 24;
        ABFrag a, b;
        const unsigned* ap = qbu + (rowQ0 + mt * 16 + mr) * 128 + h * 16;
        const unsigned* bp = kbu + ((long)s * R_ + nt * 16 + mr) * 128 + h * 16;
        #pragma unroll
        for (int j = 0; j < 8; ++j) {
            a.u[j] = ap[(j & 3) + 8 * (j >> 2) + 4 * half];
            b.u[j] = bp[(j & 3) + 8 * (j >> 2) + 4 * half];
        }
        v8f c = {};
        c = __builtin_amdgcn_wmma_f32_16x16x32_bf16(false, a.v, false, b.v, (short)0, c, false, false);
        #pragma unroll
        for (int i = 0; i < 8; ++i) {
            int m = mt * 16 + i + 8 * half;                  // local q row 0..63
            int kcol = nt * 16 + mr;                         // key index 0..383
            sc[m * SCS + kcol] = c[i]
                + biasb[(long)h * (R_ * R_) + (long)(q0 + m) * R_ + kcol]
                + mask[(long)s * R_ + kcol];
        }
    }
    __syncthreads();

    // ---- Phase 2: softmax over k (64 rows, 4 threads/row, 96 elems each)
    {
        int row = tid >> 2, part = tid & 3;
        float* srow = sc + row * SCS + part * 96;
        float mx = -3.0e38f;
        for (int i = 0; i < 96; ++i) mx = fmaxf(mx, srow[i]);
        red[tid] = mx; __syncthreads();
        if (part == 0)
            red[tid] = fmaxf(fmaxf(red[tid], red[tid + 1]), fmaxf(red[tid + 2], red[tid + 3]));
        __syncthreads();
        mx = red[row << 2];
        float sum = 0.0f;
        for (int i = 0; i < 96; ++i) { float e = __expf(srow[i] - mx); srow[i] = e; sum += e; }
        __syncthreads();
        red[tid] = sum; __syncthreads();
        if (part == 0)
            red[tid] = red[tid] + red[tid + 1] + red[tid + 2] + red[tid + 3];
        __syncthreads();
        float inv = 1.0f / red[row << 2];
        unsigned short* prow = lP + row * PST + part * 96;
        for (int i = 0; i < 96; ++i) prow[i] = bf16raw(srow[i] * inv);
    }
    __syncthreads();

    // ---- Phase 3: O = P * V  (M=64 -> 4 mt, N=32 -> 2 nt, K=384 -> 12 steps)
    {
        int mt = widx >> 1, nt = widx & 1;
        v8f acc = {};
        const unsigned* lPu = (const unsigned*)lP;           // P row = PST/2 dwords
        const unsigned* vp  = vtu + ((long)s * TOT + h * HD + nt * 16 + mr) * 192;
        for (int ks = 0; ks < 12; ++ks) {
            ABFrag a, b;
            const unsigned* ap = lPu + (mt * 16 + mr) * (PST / 2) + ks * 16;
            #pragma unroll
            for (int j = 0; j < 8; ++j) {
                a.u[j] = ap[(j & 3) + 8 * (j >> 2) + 4 * half];
                b.u[j] = vp[ks * 16 + (j & 3) + 8 * (j >> 2) + 4 * half];
            }
            acc = __builtin_amdgcn_wmma_f32_16x16x32_bf16(false, a.v, false, b.v, (short)0, acc, false, false);
        }
        #pragma unroll
        for (int i = 0; i < 8; ++i) {
            long srr = rowQ0 + mt * 16 + i + 8 * half;       // global sr
            int  e   = h * HD + nt * 16 + mr;
            float g = bf16tof(gb[srr * TOT + e]);
            og[srr * TOT + e] = bf16raw(acc[i] * g);
        }
    }
}

// ---------------------------------------------------------------------------
// Kernel 6: output projection out = og * Wo^T (f32 out). grid (768, 4)
// ---------------------------------------------------------------------------
__global__ __launch_bounds__(256)
void outproj_kernel(const unsigned* __restrict__ ogu, const unsigned* __restrict__ wo_u,
                    float* __restrict__ out) {
    __shared__ unsigned lA[2][64 * LAS];
    __shared__ unsigned lB[2][64 * LAS];
    GEMM_PROLOG();
    const int rbase = blockIdx.x * 64, nbase = blockIdx.y * 64;
    GEMM_CORE(ogu, (unsigned)SR, wo_u, 256u, rbase, nbase);

    #pragma unroll
    for (int i = 0; i < 8; ++i) {
        int gm0 = rbase + mt0 * 16 + i + 8 * half;
        int gm1 = rbase + mt1 * 16 + i + 8 * half;
        out[(long)gm0 * DIN + nbase + nt0 * 16 + mr] = acc0[i];
        out[(long)gm1 * DIN + nbase + nt1 * 16 + mr] = acc1[i];
    }
}

// ---------------------------------------------------------------------------
extern "C" void kernel_launch(void* const* d_in, const int* in_sizes, int n_in,
                              void* d_out, int out_size, void* d_ws, size_t ws_size,
                              hipStream_t stream) {
    (void)in_sizes; (void)n_in; (void)out_size; (void)ws_size;

    const float* m     = (const float*)d_in[0];
    const float* z     = (const float*)d_in[1];
    const float* mask  = (const float*)d_in[2];
    const float* ln_mw = (const float*)d_in[3];
    const float* ln_mb = (const float*)d_in[4];
    const float* ln_zw = (const float*)d_in[5];
    const float* ln_zb = (const float*)d_in[6];
    const float* Wz    = (const float*)d_in[7];
    const float* Wq    = (const float*)d_in[8];
    const float* Wk    = (const float*)d_in[9];
    const float* Wv    = (const float*)d_in[10];
    const float* Wg    = (const float*)d_in[11];
    const float* bg    = (const float*)d_in[12];
    const float* Wo    = (const float*)d_in[13];
    float* out = (float*)d_out;

    char* base = (char*)d_ws;
    const size_t SZ = (size_t)SR * TOT * 2;               // 25,165,824 bytes (bf16 mat)
    unsigned short* mn  = (unsigned short*)(base);
    unsigned short* qb  = (unsigned short*)(base + SZ);
    unsigned short* kb  = (unsigned short*)(base + 2 * SZ);
    unsigned short* vt  = (unsigned short*)(base + 3 * SZ);
    unsigned short* gb  = (unsigned short*)(base + 4 * SZ);
    unsigned short* og  = (unsigned short*)(base + 5 * SZ);
    float*          bias= (float*)(base + 6 * SZ);        // 8*384*384 f32
    unsigned short* wbf = (unsigned short*)(base + 6 * SZ + (size_t)H_ * R_ * R_ * 4);

    // 1) weights -> bf16
    convw_kernel<<<dim3(256, 5), 256, 0, stream>>>(Wq, Wk, Wv, Wg, Wo, wbf);
    // 2) layernorm(m) -> bf16
    ln_m_kernel<<<SR, 256, 0, stream>>>(m, ln_mw, ln_mb, mn);
    // 3) pair bias
    pair_bias_kernel<<<R_ * R_, 128, 0, stream>>>(z, ln_zw, ln_zb, Wz, bias);
    // 4) QKVG projections (WMMA + TDM double-buffer)
    proj_kernel<<<dim3(SR / 64, TOT / 64, 4), 256, 0, stream>>>(
        (const unsigned*)mn, (const unsigned*)wbf, bg, qb, kb, vt, gb);
    // 5) attention (WMMA)
    size_t smem = (size_t)64 * SCS * 4 + (size_t)64 * PST * 2 + 256 * 4;  // 151,552 B
    attn_kernel<<<dim3(R_ / 64, H_, S_), 256, smem, stream>>>(
        (const unsigned*)qb, (const unsigned*)kb, (const unsigned*)vt,
        gb, bias, mask, og);
    // 6) output projection (WMMA + TDM double-buffer)
    outproj_kernel<<<dim3(SR / 64, DIN / 64), 256, 0, stream>>>(
        (const unsigned*)og, (const unsigned*)(wbf + (size_t)4 * 65536), out);
}
